// NonLocalLayer_33320356282878
// MI455X (gfx1250) — compile-verified, matrix operand order checked by reading, not run on hardware
//
#include <hip/hip_runtime.h>
#include <hip/hip_bf16.h>

typedef __attribute__((ext_vector_type(16))) _Float16 v16h;
typedef __attribute__((ext_vector_type(8)))  float    v8f;

#define C_ST   2048
#define C_LT   2048
#define LAT    512
#define S_LT   512
#define NBATCH 128
#define LN_EPS 1e-5f
#define INV_SQRT_LAT 0.04419417382415922f   // 1/sqrt(512)

__device__ __forceinline__ float wave_reduce_sum(float v) {
#pragma unroll
  for (int m = 16; m > 0; m >>= 1) v += __shfl_xor(v, m, 32);
  return v;
}
__device__ __forceinline__ float wave_reduce_max(float v) {
#pragma unroll
  for (int m = 16; m > 0; m >>= 1) v = fmaxf(v, __shfl_xor(v, m, 32));
  return v;
}

__device__ __forceinline__ void pack8(v16h& d, int base, float4 lo, float4 hi) {
  d[base + 0] = (_Float16)lo.x; d[base + 1] = (_Float16)lo.y;
  d[base + 2] = (_Float16)lo.z; d[base + 3] = (_Float16)lo.w;
  d[base + 4] = (_Float16)hi.x; d[base + 5] = (_Float16)hi.y;
  d[base + 6] = (_Float16)hi.z; d[base + 7] = (_Float16)hi.w;
}

// ---------------------------------------------------------------------------
// One 16x16 output tile per wave32 via v_wmma_f32_16x16x32_f16.
// C[tm*16+m, tn*16+nn] = sum_k A[m,k] * B[nn,k] (+ biasN[nn])
// A: (m,k) at A[m*sAm + k] (K contiguous -> b128 loads).
// B: (nn,k) at B[nn*sBn + k*sBk]; BK1 => sBk==1 -> b128 loads.
// Operand layout (ISA 7.12.2, 16-bit 16x32): lane<16 -> row=lane,
// K in {0..7}+{16..23}; lane>=16 -> row=lane-16, K in {8..15}+{24..31}.
// C/D: VGPR j -> m = j + 8*(lane>=16), n = lane&15.
// ---------------------------------------------------------------------------
template<int BK1>
__global__ __launch_bounds__(32)
void gemm16_f16wmma(const float* __restrict__ A, long sAm,
                    const float* __restrict__ B, long sBn, long sBk,
                    float* __restrict__ C, long sCm,
                    const float* __restrict__ biasN,
                    int K, int ntilesN) {
  const int lane     = threadIdx.x & 31;
  const int half_sel = lane >> 4;
  const int row      = lane & 15;
  const int tm = blockIdx.x / ntilesN;
  const int tn = blockIdx.x % ntilesN;
  const float* Ab = A + (long)tm * 16 * sAm;
  const float* Bb = B + (long)tn * 16 * sBn;

  v8f acc = {};
  for (int k0 = 0; k0 < K; k0 += 32) {
    const int kb = k0 + (half_sel << 3);
    v16h a, b;
    {
      const float* ap = Ab + (long)row * sAm + kb;
      pack8(a, 0, *(const float4*)(ap),      *(const float4*)(ap + 4));
      pack8(a, 8, *(const float4*)(ap + 16), *(const float4*)(ap + 20));
    }
    if (BK1) {
      const float* bp = Bb + (long)row * sBn + kb;
      pack8(b, 0, *(const float4*)(bp),      *(const float4*)(bp + 4));
      pack8(b, 8, *(const float4*)(bp + 16), *(const float4*)(bp + 20));
    } else {
#pragma unroll
      for (int i = 0; i < 16; ++i) {
        const int kk = kb + (i & 7) + ((i >> 3) << 4);
        b[i] = (_Float16)Bb[(long)row * sBn + (long)kk * sBk];
      }
    }
    acc = __builtin_amdgcn_wmma_f32_16x16x32_f16(false, a, false, b,
                                                 (short)0, acc, false, false);
  }

  const float bias = biasN ? biasN[tn * 16 + row] : 0.0f;
  float* Cb = C + (long)tm * 16 * sCm + tn * 16;
#pragma unroll
  for (int j = 0; j < 8; ++j) {
    const int m = j + half_sel * 8;
    Cb[(long)m * sCm + row] = acc[j] + bias;
  }
}

// ---------------------------------------------------------------------------
// Fused attention core, one block per batch (folded non-local form):
//   scores[t] = u . lt[:,t] + theta.b_lt
//   p = softmax(scores / sqrt(LAT))
//   v[c] = lt[c,:] . p          (second pass hits L2: lt[n] is 4 MB)
// Pass A: 4 c-slices x 128 t-quads, b128 loads, LDS partial combine.
// ---------------------------------------------------------------------------
__global__ __launch_bounds__(512)
void attn_kernel(const float* __restrict__ lt,    // [128, 2048, 512]
                 const float* __restrict__ U,     // [128, 2048]
                 const float* __restrict__ Theta, // [128, 512]
                 const float* __restrict__ b_lt,  // [512]
                 float* __restrict__ V) {         // [128, 2048]
  __shared__ float s_u[C_LT];
  __shared__ float s_part[4 * S_LT];   // [slice][t]
  __shared__ float s_p[S_LT];
  __shared__ float s_red[16];
  const int n    = blockIdx.x;
  const int tid  = threadIdx.x;
  const int lane = tid & 31;
  const int wid  = tid >> 5;
  const float* ltn = lt + (long)n * C_LT * S_LT;

  ((float4*)s_u)[tid] = ((const float4*)(U + (long)n * C_LT))[tid];

  // beta = dot(Theta[n,:], b_lt)
  float bp = Theta[(long)n * LAT + tid] * b_lt[tid];
  bp = wave_reduce_sum(bp);
  if (lane == 0) s_red[wid] = bp;
  __syncthreads();
  {
    float r = (tid < 16) ? s_red[tid] : 0.0f;
    r = wave_reduce_sum(r);
    if (tid == 0) s_red[0] = r;
  }
  __syncthreads();
  const float beta = s_red[0];

  // ---- pass A: partial scores; thread owns t-quad (tid&127), c-slice (tid>>7)
  const int tq    = tid & 127;
  const int slice = tid >> 7;
  const float* ltq = ltn + tq * 4;
  float4 acc4 = make_float4(0.f, 0.f, 0.f, 0.f);
#pragma unroll 1
  for (int c0 = 0; c0 < 512; c0 += 8) {
    const int c = slice * 512 + c0;
    int cpre = c + 64; if (cpre > C_LT - 1) cpre = C_LT - 1;
    __builtin_prefetch(ltq + (long)cpre * S_LT, 0, 1);   // global_prefetch_b8
#pragma unroll
    for (int cc = 0; cc < 8; ++cc) {
      const float  uc = s_u[c + cc];
      const float4 x  = *(const float4*)(ltq + (long)(c + cc) * S_LT);
      acc4.x = fmaf(uc, x.x, acc4.x);
      acc4.y = fmaf(uc, x.y, acc4.y);
      acc4.z = fmaf(uc, x.z, acc4.z);
      acc4.w = fmaf(uc, x.w, acc4.w);
    }
  }
  ((float4*)s_part)[slice * 128 + tq] = acc4;
  __syncthreads();

  const float sc = s_part[0 * S_LT + tid] + s_part[1 * S_LT + tid] +
                   s_part[2 * S_LT + tid] + s_part[3 * S_LT + tid];
  const float score = (sc + beta) * INV_SQRT_LAT;
  __syncthreads();

  // ---- softmax over 512 ----
  float mx = wave_reduce_max(score);
  if (lane == 0) s_red[wid] = mx;
  __syncthreads();
  {
    float r = (tid < 16) ? s_red[tid] : -__builtin_inff();
    r = wave_reduce_max(r);
    if (tid == 0) s_red[0] = r;
  }
  __syncthreads();
  const float smax = s_red[0];
  __syncthreads();
  const float e = __expf(score - smax);
  float ssum = wave_reduce_sum(e);
  if (lane == 0) s_red[wid] = ssum;
  __syncthreads();
  {
    float r = (tid < 16) ? s_red[tid] : 0.0f;
    r = wave_reduce_sum(r);
    if (tid == 0) s_red[0] = r;
  }
  __syncthreads();
  s_p[tid] = e * (1.0f / s_red[0]);
  __syncthreads();

  // ---- pass B: v[c] = lt[c,:] . p  (b128 loads; rows are L2 resident) ----
  for (int c = wid; c < C_LT; c += 16) {
    const float* rowp = ltn + (long)c * S_LT;
    float a2 = 0.0f;
#pragma unroll
    for (int tt = 0; tt < S_LT; tt += 128) {
      const float4 x  = *(const float4*)(rowp + tt + lane * 4);
      const float4 pp = *(const float4*)(s_p  + tt + lane * 4);
      a2 = fmaf(x.x, pp.x, a2);
      a2 = fmaf(x.y, pp.y, a2);
      a2 = fmaf(x.z, pp.z, a2);
      a2 = fmaf(x.w, pp.w, a2);
    }
    a2 = wave_reduce_sum(a2);
    if (lane == 0) V[(long)n * C_LT + c] = a2;
  }
}

// ---------------------------------------------------------------------------
// LayerNorm over the channel dim (2048) per batch row (b128 loads/stores).
// ---------------------------------------------------------------------------
__global__ __launch_bounds__(256)
void ln_kernel(const float* __restrict__ X, const float* __restrict__ ln_w,
               const float* __restrict__ ln_b, float* __restrict__ out) {
  __shared__ float s_red[8];
  const int n = blockIdx.x, tid = threadIdx.x;
  const int lane = tid & 31, wid = tid >> 5;
  const float4* xr4 = (const float4*)(X + (long)n * C_LT);
  const float4 xa = xr4[tid], xb = xr4[tid + 256];
  float s = xa.x + xa.y + xa.z + xa.w + xb.x + xb.y + xb.z + xb.w;
  s = wave_reduce_sum(s);
  if (lane == 0) s_red[wid] = s;
  __syncthreads();
  {
    float r = (tid < 8) ? s_red[tid] : 0.0f;
    r = wave_reduce_sum(r);
    if (tid == 0) s_red[0] = r;
  }
  __syncthreads();
  const float mean = s_red[0] * (1.0f / C_LT);
  __syncthreads();
  float sq = 0.0f;
  {
    float d;
    d = xa.x - mean; sq = fmaf(d, d, sq);
    d = xa.y - mean; sq = fmaf(d, d, sq);
    d = xa.z - mean; sq = fmaf(d, d, sq);
    d = xa.w - mean; sq = fmaf(d, d, sq);
    d = xb.x - mean; sq = fmaf(d, d, sq);
    d = xb.y - mean; sq = fmaf(d, d, sq);
    d = xb.z - mean; sq = fmaf(d, d, sq);
    d = xb.w - mean; sq = fmaf(d, d, sq);
  }
  sq = wave_reduce_sum(sq);
  if (lane == 0) s_red[wid] = sq;
  __syncthreads();
  {
    float r = (tid < 8) ? s_red[tid] : 0.0f;
    r = wave_reduce_sum(r);
    if (tid == 0) s_red[0] = r;
  }
  __syncthreads();
  const float rstd = rsqrtf(s_red[0] * (1.0f / C_LT) + LN_EPS);
  const float4* w4 = (const float4*)ln_w;
  const float4* b4 = (const float4*)ln_b;
  float4* o4 = (float4*)(out + (long)n * C_LT);
  const float4 wa = w4[tid], wb = w4[tid + 256];
  const float4 ba = b4[tid], bb = b4[tid + 256];
  float4 oa, ob;
  oa.x = (xa.x - mean) * rstd * wa.x + ba.x;
  oa.y = (xa.y - mean) * rstd * wa.y + ba.y;
  oa.z = (xa.z - mean) * rstd * wa.z + ba.z;
  oa.w = (xa.w - mean) * rstd * wa.w + ba.w;
  ob.x = (xb.x - mean) * rstd * wb.x + bb.x;
  ob.y = (xb.y - mean) * rstd * wb.y + bb.y;
  ob.z = (xb.z - mean) * rstd * wb.z + bb.z;
  ob.w = (xb.w - mean) * rstd * wb.w + bb.w;
  o4[tid]       = oa;
  o4[tid + 256] = ob;
}

extern "C" void kernel_launch(void* const* d_in, const int* in_sizes, int n_in,
                              void* d_out, int out_size, void* d_ws, size_t ws_size,
                              hipStream_t stream) {
  const float* st    = (const float*)d_in[0];   // [128,2048]
  const float* lt    = (const float*)d_in[1];   // [128,2048,512]
  const float* w_st  = (const float*)d_in[2];   // [512,2048]
  const float* b_st  = (const float*)d_in[3];   // [512]
  const float* w_lt  = (const float*)d_in[4];   // [512,2048]
  const float* b_lt  = (const float*)d_in[5];   // [512]
  const float* w_g   = (const float*)d_in[6];   // [512,2048]
  const float* b_g   = (const float*)d_in[7];   // [512]
  const float* w_out = (const float*)d_in[8];   // [2048,512]
  const float* b_out = (const float*)d_in[9];   // [2048]
  const float* ln_w  = (const float*)d_in[10];  // [2048]
  const float* ln_b  = (const float*)d_in[11];  // [2048]
  float* out = (float*)d_out;

  float* ws    = (float*)d_ws;
  float* Theta = ws;                          // 128*512
  float* U     = Theta + NBATCH * LAT;        // 128*2048
  float* V     = U     + NBATCH * C_LT;       // 128*2048
  float* OutL  = V     + NBATCH * C_LT;       // 128*512
  float* Xpre  = OutL  + NBATCH * LAT;        // 128*2048  (total ~3.5 MB)

  // 1) Theta[n,l] = sum_c st[n,c] * w_st[l,c] + b_st[l]          (K=2048)
  gemm16_f16wmma<1><<<dim3((NBATCH / 16) * (LAT / 16)), 32, 0, stream>>>(
      st, C_ST, w_st, C_ST, 1, Theta, LAT, b_st, C_ST, LAT / 16);

  // 2) U[n,c] = sum_l Theta[n,l] * w_lt[l,c]                     (K=512)
  //    B element (c,l) = w_lt[l*2048 + c]  -> sBn=1, sBk=2048
  gemm16_f16wmma<0><<<dim3((NBATCH / 16) * (C_LT / 16)), 32, 0, stream>>>(
      Theta, LAT, w_lt, 1, C_LT, U, C_LT, nullptr, LAT, C_LT / 16);

  // 3) folded attention: scores -> softmax -> v   (2 passes over lt, L2-fused)
  attn_kernel<<<dim3(NBATCH), 512, 0, stream>>>(lt, U, Theta, b_lt, V);

  // 4) OutL[n,l] = sum_c V[n,c] * w_g[l,c] + b_g[l]              (K=2048)
  gemm16_f16wmma<1><<<dim3((NBATCH / 16) * (LAT / 16)), 32, 0, stream>>>(
      V, C_LT, w_g, C_LT, 1, OutL, LAT, b_g, C_LT, LAT / 16);

  // 5) Xpre[n,c] = sum_l OutL[n,l] * w_out[c,l] + b_out[c]       (K=512)
  gemm16_f16wmma<1><<<dim3((NBATCH / 16) * (C_LT / 16)), 32, 0, stream>>>(
      OutL, LAT, w_out, LAT, 1, Xpre, C_LT, b_out, LAT, C_LT / 16);

  // 6) LayerNorm over channels -> output
  ln_kernel<<<dim3(NBATCH), 256, 0, stream>>>(Xpre, ln_w, ln_b, out);
}
